// PatchNCELoss_28132035789367
// MI455X (gfx1250) — compile-verified, matrix operand order checked by reading, not run on hardware
//
#include <hip/hip_runtime.h>

#define DIMF 128
#define TEMP_INV 14.285714285714286f             // 1 / 0.07
#define LOG2E 1.4426950408889634f
#define SCALE_A (TEMP_INV * LOG2E)               // folded into A-side f16 data
#define EPS 1e-12f
#define CSPLIT 4                                 // column-range splits per row block
#define MROWS 64                                 // rows per block (4 x 16-row tiles)

typedef _Float16 v16h __attribute__((ext_vector_type(16)));
typedef _Float16 v8h  __attribute__((ext_vector_type(8)));
typedef _Float16 h4   __attribute__((ext_vector_type(4)));
typedef float    v8f  __attribute__((ext_vector_type(8)));

// ---------------------------------------------------------------------------
// Kernel 1: per-row L2 normalize (fp32 -> f16), compute positive logits and
// branch flags. A-side matrices (fvn/fin) are additionally scaled by
// (1/T)*log2(e) so the WMMA emits exp2-ready logits. One wave32 per row.
// ---------------------------------------------------------------------------
__global__ void patchnce_prep(const float* __restrict__ fv,
                              const float* __restrict__ fi,
                              const float* __restrict__ v,
                              const float* __restrict__ vi,
                              _Float16* __restrict__ fvn,
                              _Float16* __restrict__ fin,
                              _Float16* __restrict__ vn,
                              _Float16* __restrict__ vin,
                              float* __restrict__ posOut,
                              int* __restrict__ usevOut) {
  const int row  = blockIdx.x * 8 + (threadIdx.x >> 5);
  const int lane = threadIdx.x & 31;
  const int base = row * DIMF + lane * 4;

  float4 a = *(const float4*)(fv + base);
  float4 b = *(const float4*)(fi + base);
  float4 c = *(const float4*)(v  + base);
  float4 d = *(const float4*)(vi + base);

  float ssa = a.x * a.x + a.y * a.y + a.z * a.z + a.w * a.w;
  float ssb = b.x * b.x + b.y * b.y + b.z * b.z + b.w * b.w;
  float ssc = c.x * c.x + c.y * c.y + c.z * c.z + c.w * c.w;
  float ssd = d.x * d.x + d.y * d.y + d.z * d.z + d.w * d.w;
  float dac = a.x * c.x + a.y * c.y + a.z * c.z + a.w * c.w;  // fv . v
  float dbd = b.x * d.x + b.y * d.y + b.z * d.z + b.w * d.w;  // fi . vi

#pragma unroll
  for (int off = 1; off < 32; off <<= 1) {
    ssa += __shfl_xor(ssa, off, 32);
    ssb += __shfl_xor(ssb, off, 32);
    ssc += __shfl_xor(ssc, off, 32);
    ssd += __shfl_xor(ssd, off, 32);
    dac += __shfl_xor(dac, off, 32);
    dbd += __shfl_xor(dbd, off, 32);
  }

  const float inva = 1.0f / fmaxf(sqrtf(ssa), EPS);
  const float invb = 1.0f / fmaxf(sqrtf(ssb), EPS);
  const float invc = 1.0f / fmaxf(sqrtf(ssc), EPS);
  const float invd = 1.0f / fmaxf(sqrtf(ssd), EPS);

  const float pv = dac * inva * invc;
  const float pi = dbd * invb * invd;
  const bool  uv = pv >= pi;
  if (lane == 0) {
    posOut[row]  = uv ? pv : pi;
    usevOut[row] = uv ? 1 : 0;
  }

  const float sa = inva * SCALE_A;   // A-side: fold 1/T * log2(e)
  const float sb = invb * SCALE_A;
  h4 ha = {(_Float16)(a.x * sa), (_Float16)(a.y * sa),
           (_Float16)(a.z * sa), (_Float16)(a.w * sa)};
  h4 hb = {(_Float16)(b.x * sb), (_Float16)(b.y * sb),
           (_Float16)(b.z * sb), (_Float16)(b.w * sb)};
  h4 hc = {(_Float16)(c.x * invc), (_Float16)(c.y * invc),
           (_Float16)(c.z * invc), (_Float16)(c.w * invc)};
  h4 hd = {(_Float16)(d.x * invd), (_Float16)(d.y * invd),
           (_Float16)(d.z * invd), (_Float16)(d.w * invd)};
  *(h4*)(fvn + base) = ha;
  *(h4*)(fin + base) = hb;
  *(h4*)(vn  + base) = hc;
  *(h4*)(vin + base) = hd;
}

// ---------------------------------------------------------------------------
// Kernel 2: stable partition of rows by use_v (v-rows first). Single block.
// rowOf[newIdx] = original row; nvOut[0] = count of v-rows. Also zeroes gsum.
// ---------------------------------------------------------------------------
__global__ void build_perm(const int* __restrict__ usev, int B,
                           int* __restrict__ rowOf, int* __restrict__ nvOut,
                           float* __restrict__ gsum) {
  __shared__ int cnt[256];
  const int tid = threadIdx.x;
  const int per = B / 256;
  for (int i = tid; i < B; i += 256) gsum[i] = 0.0f;
  int c = 0;
  for (int i = 0; i < per; ++i) c += usev[tid * per + i];
  cnt[tid] = c;
  __syncthreads();
  for (int off = 1; off < 256; off <<= 1) {  // inclusive Hillis-Steele scan
    int add = (tid >= off) ? cnt[tid - off] : 0;
    __syncthreads();
    cnt[tid] += add;
    __syncthreads();
  }
  const int total   = cnt[255];
  const int beforeV = (tid == 0) ? 0 : cnt[tid - 1];
  int posV = beforeV;
  int posI = total + (tid * per - beforeV);
  for (int i = 0; i < per; ++i) {
    const int r = tid * per + i;
    if (usev[r]) rowOf[posV++] = r;
    else         rowOf[posI++] = r;
  }
  if (tid == 0) nvOut[0] = total;
}

// ---------------------------------------------------------------------------
// Fragment loader matching the CDNA5 16-bit A/B WMMA layout (ISA 7.12.2):
// lane L holds row (L&15); with hi = L>>4, its 16 f16 are
//   K = kBase + hi*8 + [0..7]  and  K = kBase + hi*8 + 16 + [0..7]
// -> two contiguous b128 loads from a row-major [*, 128] matrix.
// ---------------------------------------------------------------------------
__device__ __forceinline__ v16h load_frag(const _Float16* __restrict__ p) {
  v8h lo = *(const v8h*)(p);
  v8h hi = *(const v8h*)(p + 16);
  v16h r;
#pragma unroll
  for (int i = 0; i < 8; ++i) { r[i] = lo[i]; r[i + 8] = hi[i]; }
  return r;
}

__device__ __forceinline__ void load_tile(const _Float16* __restrict__ rowp,
                                          v16h (&buf)[4]) {
#pragma unroll
  for (int c = 0; c < 4; ++c) buf[c] = load_frag(rowp + c * 32);
}

// 16 WMMAs (4 row-tiles x 4 K-chunks) on one B-fragment set + exp2 accumulate.
// A-side is pre-scaled, so each element needs only one v_exp_f32.
__device__ __forceinline__ void tile_compute(const v16h (&a0)[4],
                                             const v16h (&a1)[4],
                                             const v16h (&a2)[4],
                                             const v16h (&a3)[4],
                                             const v16h (&b)[4],
                                             const int (&orow)[4][8],
                                             int col,
                                             float (&acc)[4][8]) {
  v8f c0 = {0.f,0.f,0.f,0.f,0.f,0.f,0.f,0.f};
  v8f c1 = {0.f,0.f,0.f,0.f,0.f,0.f,0.f,0.f};
  v8f c2 = {0.f,0.f,0.f,0.f,0.f,0.f,0.f,0.f};
  v8f c3 = {0.f,0.f,0.f,0.f,0.f,0.f,0.f,0.f};
#pragma unroll
  for (int c = 0; c < 4; ++c) {
    c0 = __builtin_amdgcn_wmma_f32_16x16x32_f16(false, a0[c], false, b[c],
                                                (short)0, c0, false, false);
    c1 = __builtin_amdgcn_wmma_f32_16x16x32_f16(false, a1[c], false, b[c],
                                                (short)0, c1, false, false);
    c2 = __builtin_amdgcn_wmma_f32_16x16x32_f16(false, a2[c], false, b[c],
                                                (short)0, c2, false, false);
    c3 = __builtin_amdgcn_wmma_f32_16x16x32_f16(false, a3[c], false, b[c],
                                                (short)0, c3, false, false);
  }
#pragma unroll
  for (int r = 0; r < 8; ++r) {
    acc[0][r] += (orow[0][r] == col) ? 0.0f : __builtin_amdgcn_exp2f(c0[r]);
    acc[1][r] += (orow[1][r] == col) ? 0.0f : __builtin_amdgcn_exp2f(c1[r]);
    acc[2][r] += (orow[2][r] == col) ? 0.0f : __builtin_amdgcn_exp2f(c2[r]);
    acc[3][r] += (orow[3][r] == col) ? 0.0f : __builtin_amdgcn_exp2f(c3[r]);
  }
}

// ---------------------------------------------------------------------------
// Kernel 3: fused WMMA GEMM + online exp-sum over a 64-row x (B/CSPLIT)-col
// panel per block. Row-partitioned order makes (almost) all blocks single-
// branch: 16 WMMAs per 8 b128 loads, double-buffered. Partial per-row sums
// go to gsum[] (permuted index) via LDS then global atomics.
// ---------------------------------------------------------------------------
__global__ void patchnce_gemm(const _Float16* __restrict__ fvn,
                              const _Float16* __restrict__ fin,
                              const _Float16* __restrict__ vn,
                              const _Float16* __restrict__ vin,
                              const int* __restrict__ rowOf,
                              const int* __restrict__ nvPtr,
                              float* __restrict__ gsum,
                              int B) {
  const int rowBlk  = blockIdx.x >> 2;          // / CSPLIT
  const int cs      = blockIdx.x & (CSPLIT - 1);
  const int rowBase = rowBlk * MROWS;           // position in PERMUTED order
  const int tid  = threadIdx.x;
  const int wave = tid >> 5;
  const int lane = tid & 31;
  const int rIn  = lane & 15;
  const int hi   = lane >> 4;
  const int nv   = nvPtr[0];

  const int nTiles   = B >> 4;
  const int tilesPer = nTiles / CSPLIT;
  const int tStart   = cs * tilesPer;
  const int tEnd     = tStart + tilesPer;

  __shared__ float ssum[MROWS];
  __shared__ int   sorig[MROWS];
  if (tid < MROWS) {
    ssum[tid]  = 0.0f;
    sorig[tid] = rowOf[rowBase + tid];
  }
  __syncthreads();

  const bool allV = (rowBase + MROWS) <= nv;
  const bool allI = rowBase >= nv;

  if (allV || allI) {
    // ---------------- homogeneous fast path: one branch only ----------------
    const _Float16* Amat = allV ? fvn : fin;
    const _Float16* Bmat = allV ? vn  : vin;

    v16h a0[4], a1[4], a2[4], a3[4];
    load_tile(Amat + (size_t)sorig[ 0 + rIn] * DIMF + hi * 8, a0);
    load_tile(Amat + (size_t)sorig[16 + rIn] * DIMF + hi * 8, a1);
    load_tile(Amat + (size_t)sorig[32 + rIn] * DIMF + hi * 8, a2);
    load_tile(Amat + (size_t)sorig[48 + rIn] * DIMF + hi * 8, a3);

    int orow[4][8];
    float acc[4][8];
#pragma unroll
    for (int rt = 0; rt < 4; ++rt)
#pragma unroll
      for (int r = 0; r < 8; ++r) {
        orow[rt][r] = sorig[rt * 16 + r + 8 * hi];
        acc[rt][r]  = 0.0f;
      }

    v16h b0[4], b1[4];
    int t = tStart + wave;
    if (t < tEnd)
      load_tile(Bmat + (size_t)((t << 4) + rIn) * DIMF + hi * 8, b0);

    for (; t < tEnd; t += 16) {
      const bool has2 = (t + 8) < tEnd;
      if (has2)
        load_tile(Bmat + (size_t)(((t + 8) << 4) + rIn) * DIMF + hi * 8, b1);
      tile_compute(a0, a1, a2, a3, b0, orow, (t << 4) + rIn, acc);
      if ((t + 16) < tEnd)
        load_tile(Bmat + (size_t)(((t + 16) << 4) + rIn) * DIMF + hi * 8, b0);
      if (has2)
        tile_compute(a0, a1, a2, a3, b1, orow, ((t + 8) << 4) + rIn, acc);
    }

#pragma unroll
    for (int rt = 0; rt < 4; ++rt)
#pragma unroll
      for (int r = 0; r < 8; ++r)
#pragma unroll
        for (int off = 1; off < 16; off <<= 1)
          acc[rt][r] += __shfl_xor(acc[rt][r], off, 32);
    if (rIn == 0) {
#pragma unroll
      for (int rt = 0; rt < 4; ++rt)
#pragma unroll
        for (int r = 0; r < 8; ++r)
          atomicAdd(&ssum[rt * 16 + r + 8 * hi], acc[rt][r]);
    }
  } else {
    // ------- mixed row block (<= CSPLIT blocks): per-row-tile fallback ------
    for (int rt = 0; rt < 4; ++rt) {
      const int aorig = sorig[rt * 16 + rIn];
      v16h afv[4], afi[4];
      load_tile(fvn + (size_t)aorig * DIMF + hi * 8, afv);
      load_tile(fin + (size_t)aorig * DIMF + hi * 8, afi);

      int orow[8];
      bool uv[8];
      float acc[8];
#pragma unroll
      for (int r = 0; r < 8; ++r) {
        orow[r] = sorig[rt * 16 + r + 8 * hi];
        uv[r]   = (rowBase + rt * 16 + r + 8 * hi) < nv;
        acc[r]  = 0.0f;
      }

      for (int t = tStart + wave; t < tEnd; t += 8) {
        const int colBase = t << 4;
        v16h bv[4], bi[4];
        load_tile(vn  + (size_t)(colBase + rIn) * DIMF + hi * 8, bv);
        load_tile(vin + (size_t)(colBase + rIn) * DIMF + hi * 8, bi);
        v8f cv = {0.f,0.f,0.f,0.f,0.f,0.f,0.f,0.f};
        v8f ci = {0.f,0.f,0.f,0.f,0.f,0.f,0.f,0.f};
#pragma unroll
        for (int c = 0; c < 4; ++c) {
          cv = __builtin_amdgcn_wmma_f32_16x16x32_f16(false, afv[c], false,
                                                      bv[c], (short)0, cv,
                                                      false, false);
          ci = __builtin_amdgcn_wmma_f32_16x16x32_f16(false, afi[c], false,
                                                      bi[c], (short)0, ci,
                                                      false, false);
        }
        const int col = colBase + rIn;
#pragma unroll
        for (int r = 0; r < 8; ++r) {
          const float s = uv[r] ? cv[r] : ci[r];
          acc[r] += (orow[r] == col) ? 0.0f : __builtin_amdgcn_exp2f(s);
        }
      }
#pragma unroll
      for (int r = 0; r < 8; ++r)
#pragma unroll
        for (int off = 1; off < 16; off <<= 1)
          acc[r] += __shfl_xor(acc[r], off, 32);
      if (rIn == 0) {
#pragma unroll
        for (int r = 0; r < 8; ++r)
          atomicAdd(&ssum[rt * 16 + r + 8 * hi], acc[r]);
      }
    }
  }

  __syncthreads();
  if (tid < MROWS)
    atomicAdd(&gsum[rowBase + tid], ssum[tid]);
}

// ---------------------------------------------------------------------------
// Kernel 4: finalize. loss = log(exp(pos/T) + sum_{k!=b} exp(S/T)) - pos/T
// ---------------------------------------------------------------------------
__global__ void patchnce_finalize(const float* __restrict__ gsum,
                                  const float* __restrict__ pos,
                                  const int* __restrict__ rowOf,
                                  float* __restrict__ out, int B) {
  const int p = blockIdx.x * 256 + threadIdx.x;
  if (p < B) {
    const int orig = rowOf[p];
    const float pp = pos[orig] * TEMP_INV;
    out[orig] = __logf(__expf(pp) + gsum[p]) - pp;
  }
}

// ---------------------------------------------------------------------------
extern "C" void kernel_launch(void* const* d_in, const int* in_sizes, int n_in,
                              void* d_out, int out_size, void* d_ws, size_t ws_size,
                              hipStream_t stream) {
  const float* fv = (const float*)d_in[0];
  const float* fi = (const float*)d_in[1];
  const float* v  = (const float*)d_in[2];
  const float* vi = (const float*)d_in[3];
  const int B = in_sizes[0] / DIMF;  // 8192

  // Workspace: 4 x [B,128] f16 (8 MB) + pos[B] + usev[B] + rowOf[B] + gsum[B] + nv
  const size_t featElems = (size_t)B * DIMF;
  _Float16* fvn = (_Float16*)d_ws;
  _Float16* fin = fvn + featElems;
  _Float16* vn  = fin + featElems;
  _Float16* vin = vn  + featElems;
  float* pos   = (float*)(vin + featElems);
  int*   usev  = (int*)(pos + B);
  int*   rowOf = usev + B;
  float* gsum  = (float*)(rowOf + B);
  int*   nv    = (int*)(gsum + B);

  patchnce_prep<<<B / 8, 256, 0, stream>>>(fv, fi, v, vi, fvn, fin, vn, vin,
                                           pos, usev);
  build_perm<<<1, 256, 0, stream>>>(usev, B, rowOf, nv, gsum);
  patchnce_gemm<<<(B / MROWS) * CSPLIT, 256, 0, stream>>>(fvn, fin, vn, vin,
                                                          rowOf, nv, gsum, B);
  patchnce_finalize<<<(B + 255) / 256, 256, 0, stream>>>(gsum, pos, rowOf,
                                                         (float*)d_out, B);
}
